// EncoderLayer_71683004170482
// MI455X (gfx1250) — matrix operand order repaired and checked by reading, not compiled, not run on hardware
//
#include <hip/hip_runtime.h>
#include <hip/hip_bf16.h>

#define D_MODEL 768
#define SEQ     2048
#define BATCH   2
#define NHEADS  12
#define DK      64
#define DFF     3072
#define ROWS    (BATCH * SEQ)   // 4096

typedef __attribute__((ext_vector_type(8)))  float          v8f;
typedef __attribute__((ext_vector_type(16))) __bf16         v16bf;
typedef __attribute__((ext_vector_type(16))) unsigned short v16u;
typedef __attribute__((ext_vector_type(8)))  unsigned short v8u;
typedef __attribute__((ext_vector_type(4)))  unsigned short v4u;
typedef int v4i __attribute__((vector_size(16)));

union FragBF {
    v16u u;
    v16bf b;
    v8u  h[2];
};

__device__ __forceinline__ unsigned short f2bf(float f) {
    union { float f; unsigned int u; } x; x.f = f;
    unsigned int u = x.u;
    unsigned int r = u + 0x7fffu + ((u >> 16) & 1u);   // round-to-nearest-even
    return (unsigned short)(r >> 16);
}

// ---- CDNA5 async global->LDS copy (ASYNCcnt path), with sync fallback ------
#if defined(__has_builtin)
#  if __has_builtin(__builtin_amdgcn_global_load_async_to_lds_b128) && \
      __has_builtin(__builtin_amdgcn_s_wait_asynccnt)
#    define HAVE_ASYNC_LDS 1
#  endif
#endif
#ifndef HAVE_ASYNC_LDS
#  define HAVE_ASYNC_LDS 0
#endif

__device__ __forceinline__ void copy16_to_lds(unsigned short* lds, const unsigned short* g) {
#if HAVE_ASYNC_LDS
    __builtin_amdgcn_global_load_async_to_lds_b128(
        (__attribute__((address_space(1))) v4i*)g,
        (__attribute__((address_space(3))) v4i*)lds,
        0, 0);
#else
    *(v8u*)lds = *(const v8u*)g;
#endif
}

__device__ __forceinline__ void async_copy_wait() {
#if HAVE_ASYNC_LDS
    __builtin_amdgcn_s_wait_asynccnt(0);
#endif
}

// ---------------------------------------------------------------------------
// One-time fp32 -> bf16 convert (elementwise, float4 wide)
// ---------------------------------------------------------------------------
__global__ __launch_bounds__(256)
void convert_bf16_kernel(const float* __restrict__ src, unsigned short* __restrict__ dst)
{
    size_t i4 = (size_t)blockIdx.x * 256 + threadIdx.x;
    float4 f = *(const float4*)(src + i4 * 4);
    v4u p;
    p[0] = f2bf(f.x); p[1] = f2bf(f.y); p[2] = f2bf(f.z); p[3] = f2bf(f.w);
    *(v4u*)(dst + i4 * 4) = p;
}

// ---------------------------------------------------------------------------
// One-time weight transpose+convert: dst[N,K] bf16 = src[K,N] fp32
// 32x32 LDS tiles; grid (N/32, K/32), block 256 (32x8)
// ---------------------------------------------------------------------------
__global__ __launch_bounds__(256)
void transpose_bf16_kernel(const float* __restrict__ src, unsigned short* __restrict__ dst,
                           int K, int N)
{
    __shared__ float t[32][33];
    const int tx = threadIdx.x & 31;
    const int ty = threadIdx.x >> 5;          // 0..7
    const int kb = blockIdx.y * 32;
    const int nb = blockIdx.x * 32;
    #pragma unroll
    for (int i = 0; i < 4; ++i)
        t[ty + i * 8][tx] = src[(size_t)(kb + ty + i * 8) * N + nb + tx];
    __syncthreads();
    #pragma unroll
    for (int i = 0; i < 4; ++i)
        dst[(size_t)(nb + ty + i * 8) * K + kb + tx] = f2bf(t[tx][ty + i * 8]);
}

// ---------------------------------------------------------------------------
// bf16 GEMM: out = act(A[M,K] @ BT[N,K]^T + bias[N]); out is fp32 OR bf16.
// block = 256 threads (8 waves), tile 128(M) x 64(N), K-step 32.
// LDS tiles filled with GLOBAL_LOAD_ASYNC_TO_LDS_B128.
// ---------------------------------------------------------------------------
__global__ __launch_bounds__(256)
void gemm_bf16_kernel(const unsigned short* __restrict__ A,   // [M,K]  bf16
                      const unsigned short* __restrict__ BT,  // [N,K]  bf16 (pre-transposed)
                      const float* __restrict__ bias,
                      float* __restrict__ Cf, unsigned short* __restrict__ Cb,
                      int M, int N, int K, int relu)
{
    __shared__ unsigned short lA[128][32];   // 8 KB
    __shared__ unsigned short lB[64][32];    // 4 KB

    const int tid  = threadIdx.x;
    const int wave = tid >> 5;
    const int lane = tid & 31;
    const int l16  = lane & 15;
    const int sel  = lane >> 4;
    const int wr   = wave >> 1;              // 0..3
    const int wc   = wave & 1;               // 0..1
    const int rowbase = blockIdx.y * 128;
    const int colbase = blockIdx.x * 64;

    v8f acc[2][2] = {};

    for (int k0 = 0; k0 < K; k0 += 32) {
        __syncthreads();
        // A tile: 512 x 16B chunks (2 per thread), rows of 32 bf16 = 4 chunks
        #pragma unroll
        for (int i = 0; i < 2; ++i) {
            int id = i * 256 + tid;
            int r  = id >> 2;
            int c  = id & 3;
            copy16_to_lds(&lA[r][c * 8], A + (size_t)(rowbase + r) * K + k0 + c * 8);
        }
        // B tile: 256 x 16B chunks (1 per thread)
        {
            int n = tid >> 2;
            int c = tid & 3;
            copy16_to_lds(&lB[n][c * 8], BT + (size_t)(colbase + n) * K + k0 + c * 8);
        }
        async_copy_wait();
        __syncthreads();

        FragBF af[2], bf[2];
        #pragma unroll
        for (int rt = 0; rt < 2; ++rt) {     // A frag: lane=row, K {0..7,16..23}/{8..15,24..31}
            int r = wr * 32 + rt * 16 + l16;
            af[rt].h[0] = *(const v8u*)&lA[r][sel * 8];
            af[rt].h[1] = *(const v8u*)&lA[r][16 + sel * 8];
        }
        #pragma unroll
        for (int ct = 0; ct < 2; ++ct) {     // B frag: lane=col, K {0..7,16..23}/{8..15,24..31}
            int n = wc * 32 + ct * 16 + l16;
            bf[ct].h[0] = *(const v8u*)&lB[n][sel * 8];
            bf[ct].h[1] = *(const v8u*)&lB[n][16 + sel * 8];
        }
        #pragma unroll
        for (int rt = 0; rt < 2; ++rt)
            #pragma unroll
            for (int ct = 0; ct < 2; ++ct)
                acc[rt][ct] = __builtin_amdgcn_wmma_f32_16x16x32_bf16(
                    false, af[rt].b, false, bf[ct].b, (short)0, acc[rt][ct], false, false);
    }

    #pragma unroll
    for (int rt = 0; rt < 2; ++rt)
        #pragma unroll
        for (int ct = 0; ct < 2; ++ct)
            #pragma unroll
            for (int i = 0; i < 8; ++i) {
                int row = rowbase + wr * 32 + rt * 16 + i + sel * 8;
                int col = colbase + wc * 32 + ct * 16 + l16;
                float v = acc[rt][ct][i] + bias[col];
                if (relu) v = fmaxf(v, 0.0f);
                if (Cb) Cb[(size_t)row * N + col] = f2bf(v);
                else    Cf[(size_t)row * N + col] = v;
            }
}

// ---------------------------------------------------------------------------
// Fused flash attention, bf16 Q/K/V -> bf16 ctx. 8 waves x 16 query rows.
// K tile staged with async LDS copies; V tile staged transposed.
// ---------------------------------------------------------------------------
__global__ __launch_bounds__(256)
void flash_attn_kernel(const unsigned short* __restrict__ q,
                       const unsigned short* __restrict__ k,
                       const unsigned short* __restrict__ v,
                       unsigned short* __restrict__ ctx)
{
    __shared__ unsigned short lK[32][64];        // [key][dim]  4 KB
    __shared__ unsigned short lV[64][32];        // [dim][key]  4 KB (transposed)
    __shared__ unsigned short lP[8][16][32];     // per-wave P tile, 8 KB

    const int tid  = threadIdx.x;
    const int wave = tid >> 5;
    const int lane = tid & 31;
    const int l16  = lane & 15;
    const int sel  = lane >> 4;
    const int b    = blockIdx.y / NHEADS;
    const int h    = blockIdx.y % NHEADS;
    const int hoff = h * DK;
    const int qbase = blockIdx.x * 128 + wave * 16;

    // persistent Q fragments (A-layout), two K-steps of 32 dims
    FragBF qf[2];
    {
        const unsigned short* src =
            q + (size_t)(b * SEQ + qbase + l16) * D_MODEL + hoff;
        #pragma unroll
        for (int ks = 0; ks < 2; ++ks)
            #pragma unroll
            for (int c2 = 0; c2 < 2; ++c2)
                qf[ks].h[c2] = *(const v8u*)(src + ks * 32 + c2 * 16 + sel * 8);
    }

    v8f acc[4] = {};
    float m[8], l[8];
    #pragma unroll
    for (int i = 0; i < 8; ++i) { m[i] = -3.0e38f; l[i] = 0.0f; }

    const int key = tid >> 3;                   // staging role: 32 keys x 8 chunks
    const int ck  = tid & 7;

    for (int j = 0; j < SEQ; j += 32) {
        __syncthreads();
        size_t gro = (size_t)(b * SEQ + j + key) * D_MODEL + hoff + ck * 8;
        // K tile via async LDS copy
        copy16_to_lds(&lK[key][ck * 8], k + gro);
        // V tile transposed (scalar scatter)
        {
            v8u g = *(const v8u*)(v + gro);
            #pragma unroll
            for (int e = 0; e < 8; ++e) lV[ck * 8 + e][key] = g[e];
        }
        async_copy_wait();
        __syncthreads();

        // scores: two 16x16 tiles over K-dim 64
        v8f s0 = {}, s1 = {};
        #pragma unroll
        for (int ks = 0; ks < 2; ++ks) {
            FragBF kf0, kf1;
            kf0.h[0] = *(const v8u*)&lK[l16     ][ks * 32 + sel * 8];
            kf0.h[1] = *(const v8u*)&lK[l16     ][ks * 32 + 16 + sel * 8];
            kf1.h[0] = *(const v8u*)&lK[16 + l16][ks * 32 + sel * 8];
            kf1.h[1] = *(const v8u*)&lK[16 + l16][ks * 32 + 16 + sel * 8];
            s0 = __builtin_amdgcn_wmma_f32_16x16x32_bf16(false, qf[ks].b, false, kf0.b,
                                                         (short)0, s0, false, false);
            s1 = __builtin_amdgcn_wmma_f32_16x16x32_bf16(false, qf[ks].b, false, kf1.b,
                                                         (short)0, s1, false, false);
        }

        // online softmax across the 32 new columns
        #pragma unroll
        for (int i = 0; i < 8; ++i) {
            float a0 = s0[i] * 0.125f;
            float a1 = s1[i] * 0.125f;
            float mx = fmaxf(a0, a1);
            #pragma unroll
            for (int msk = 1; msk < 16; msk <<= 1)
                mx = fmaxf(mx, __shfl_xor(mx, msk, 32));
            float mn    = fmaxf(m[i], mx);
            float alpha = __expf(m[i] - mn);
            float e0    = __expf(a0 - mn);
            float e1    = __expf(a1 - mn);
            float rs    = e0 + e1;
            #pragma unroll
            for (int msk = 1; msk < 16; msk <<= 1)
                rs += __shfl_xor(rs, msk, 32);
            l[i] = l[i] * alpha + rs;
            m[i] = mn;
            #pragma unroll
            for (int c = 0; c < 4; ++c) acc[c][i] *= alpha;
            lP[wave][i + sel * 8][l16]      = f2bf(e0);
            lP[wave][i + sel * 8][16 + l16] = f2bf(e1);
        }

        // ctx += P(16x32) @ V(32x64)
        FragBF pf;
        pf.h[0] = *(const v8u*)&lP[wave][l16][sel * 8];
        pf.h[1] = *(const v8u*)&lP[wave][l16][16 + sel * 8];
        #pragma unroll
        for (int c = 0; c < 4; ++c) {
            FragBF vf;
            int n = c * 16 + l16;
            vf.h[0] = *(const v8u*)&lV[n][sel * 8];
            vf.h[1] = *(const v8u*)&lV[n][16 + sel * 8];
            acc[c] = __builtin_amdgcn_wmma_f32_16x16x32_bf16(false, pf.b, false, vf.b,
                                                             (short)0, acc[c], false, false);
        }
    }

    #pragma unroll
    for (int i = 0; i < 8; ++i) {
        float inv = 1.0f / l[i];
        int row = qbase + i + sel * 8;
        #pragma unroll
        for (int c = 0; c < 4; ++c)
            ctx[(size_t)(b * SEQ + row) * D_MODEL + hoff + c * 16 + l16] =
                f2bf(acc[c][i] * inv);
    }
}

// ---------------------------------------------------------------------------
// out = LayerNorm(x + y)*g + be ; fp32 out, optional extra bf16 copy
// ---------------------------------------------------------------------------
__global__ __launch_bounds__(256)
void add_ln_kernel(const float* __restrict__ x, const float* __restrict__ y,
                   const float* __restrict__ g, const float* __restrict__ be,
                   float* __restrict__ outf, unsigned short* __restrict__ outb)
{
    __shared__ float red1[8];
    __shared__ float red2[8];
    const int row  = blockIdx.x;
    const int tid  = threadIdx.x;
    const int wave = tid >> 5;
    const int lane = tid & 31;
    const float* xr = x + (size_t)row * D_MODEL;
    const float* yr = y + (size_t)row * D_MODEL;

    float vals[3];
    float s = 0.0f;
    #pragma unroll
    for (int i = 0; i < 3; ++i) {
        int c = tid + i * 256;
        vals[i] = xr[c] + yr[c];
        s += vals[i];
    }
    #pragma unroll
    for (int msk = 1; msk < 32; msk <<= 1) s += __shfl_xor(s, msk, 32);
    if (lane == 0) red1[wave] = s;
    __syncthreads();
    float tot = 0.0f;
    #pragma unroll
    for (int w = 0; w < 8; ++w) tot += red1[w];
    float mu = tot * (1.0f / (float)D_MODEL);

    float vs = 0.0f;
    #pragma unroll
    for (int i = 0; i < 3; ++i) { float d = vals[i] - mu; vs += d * d; }
    #pragma unroll
    for (int msk = 1; msk < 32; msk <<= 1) vs += __shfl_xor(vs, msk, 32);
    if (lane == 0) red2[wave] = vs;
    __syncthreads();
    float vtot = 0.0f;
    #pragma unroll
    for (int w = 0; w < 8; ++w) vtot += red2[w];
    float rstd = rsqrtf(vtot * (1.0f / (float)D_MODEL) + 1e-5f);

    #pragma unroll
    for (int i = 0; i < 3; ++i) {
        int c = tid + i * 256;
        float r = (vals[i] - mu) * rstd * g[c] + be[c];
        outf[(size_t)row * D_MODEL + c] = r;
        if (outb) outb[(size_t)row * D_MODEL + c] = f2bf(r);
    }
}

// ---------------------------------------------------------------------------
extern "C" void kernel_launch(void* const* d_in, const int* in_sizes, int n_in,
                              void* d_out, int out_size, void* d_ws, size_t ws_size,
                              hipStream_t stream)
{
    const float* x   = (const float*)d_in[0];
    const float* Wq  = (const float*)d_in[1];
    const float* bq  = (const float*)d_in[2];
    const float* Wk  = (const float*)d_in[3];
    const float* bk  = (const float*)d_in[4];
    const float* Wv  = (const float*)d_in[5];
    const float* bv  = (const float*)d_in[6];
    const float* Wo  = (const float*)d_in[7];
    const float* bo  = (const float*)d_in[8];
    const float* W1  = (const float*)d_in[9];
    const float* b1  = (const float*)d_in[10];
    const float* W2  = (const float*)d_in[11];
    const float* b2  = (const float*)d_in[12];
    const float* g1  = (const float*)d_in[13];
    const float* be1 = (const float*)d_in[14];
    const float* g2  = (const float*)d_in[15];
    const float* be2 = (const float*)d_in[16];
    float* out = (float*)d_out;

    const size_t NBS = (size_t)ROWS * D_MODEL;       // 4096*768
    const size_t NFF = (size_t)ROWS * DFF;           // 4096*3072
    const size_t NW  = (size_t)D_MODEL * D_MODEL;    // 768*768
    const size_t NW1 = (size_t)D_MODEL * DFF;

    // bump allocation: fp32 region first, then bf16 (u16) region
    float* fptr = (float*)d_ws;
    float* attn_f = fptr; fptr += NBS;
    float* x1f    = fptr; fptr += NBS;
    float* ff_f   = fptr; fptr += NBS;

    unsigned short* uptr = (unsigned short*)fptr;
    unsigned short* xb   = uptr; uptr += NBS;
    unsigned short* WqT  = uptr; uptr += NW;
    unsigned short* WkT  = uptr; uptr += NW;
    unsigned short* WvT  = uptr; uptr += NW;
    unsigned short* WoT  = uptr; uptr += NW;
    unsigned short* W1T  = uptr; uptr += NW1;
    unsigned short* W2T  = uptr; uptr += NW1;
    unsigned short* qb   = uptr; uptr += NBS;
    unsigned short* kb   = uptr; uptr += NBS;
    unsigned short* vb   = uptr; uptr += NBS;
    unsigned short* ctxb = uptr; uptr += NBS;
    unsigned short* x1b  = uptr; uptr += NBS;
    unsigned short* h1b  = uptr; uptr += NFF;

    dim3 blk(256);
    dim3 gD(D_MODEL / 64, ROWS / 128);               // (12, 32)
    dim3 gF(DFF / 64,     ROWS / 128);               // (48, 32)
    dim3 gA(SEQ / 128, BATCH * NHEADS);              // (16, 24)
    dim3 gLN(ROWS);

    // one-time conversions
    convert_bf16_kernel<<<dim3(NBS / 1024), blk, 0, stream>>>(x, xb);
    transpose_bf16_kernel<<<dim3(D_MODEL / 32, D_MODEL / 32), blk, 0, stream>>>(Wq, WqT, D_MODEL, D_MODEL);
    transpose_bf16_kernel<<<dim3(D_MODEL / 32, D_MODEL / 32), blk, 0, stream>>>(Wk, WkT, D_MODEL, D_MODEL);
    transpose_bf16_kernel<<<dim3(D_MODEL / 32, D_MODEL / 32), blk, 0, stream>>>(Wv, WvT, D_MODEL, D_MODEL);
    transpose_bf16_kernel<<<dim3(D_MODEL / 32, D_MODEL / 32), blk, 0, stream>>>(Wo, WoT, D_MODEL, D_MODEL);
    transpose_bf16_kernel<<<dim3(DFF / 32,     D_MODEL / 32), blk, 0, stream>>>(W1, W1T, D_MODEL, DFF);
    transpose_bf16_kernel<<<dim3(D_MODEL / 32, DFF / 32),     blk, 0, stream>>>(W2, W2T, DFF, D_MODEL);

    // QKV projections (bf16 out)
    gemm_bf16_kernel<<<gD, blk, 0, stream>>>(xb, WqT, bq, nullptr, qb, ROWS, D_MODEL, D_MODEL, 0);
    gemm_bf16_kernel<<<gD, blk, 0, stream>>>(xb, WkT, bk, nullptr, kb, ROWS, D_MODEL, D_MODEL, 0);
    gemm_bf16_kernel<<<gD, blk, 0, stream>>>(xb, WvT, bv, nullptr, vb, ROWS, D_MODEL, D_MODEL, 0);
    // fused attention -> bf16 ctx
    flash_attn_kernel<<<gA, blk, 0, stream>>>(qb, kb, vb, ctxb);
    // output projection (fp32 out for residual)
    gemm_bf16_kernel<<<gD, blk, 0, stream>>>(ctxb, WoT, bo, attn_f, nullptr, ROWS, D_MODEL, D_MODEL, 0);
    // x1 = LN(x + attn_out): fp32 + bf16 copies
    add_ln_kernel<<<gLN, blk, 0, stream>>>(x, attn_f, g1, be1, x1f, x1b);
    // FFN
    gemm_bf16_kernel<<<gF, blk, 0, stream>>>(x1b, W1T, b1, nullptr, h1b, ROWS, DFF, D_MODEL, 1);
    gemm_bf16_kernel<<<gD, blk, 0, stream>>>(h1b, W2T, b2, ff_f, nullptr, ROWS, D_MODEL, DFF, 0);
    // out = LN(x1 + ff)
    add_ln_kernel<<<gLN, blk, 0, stream>>>(x1f, ff_f, g2, be2, out, nullptr);
}
